// BEVSampling_66735201845895
// MI455X (gfx1250) — compile-verified, matrix operand order checked by reading, not run on hardware
//
#include <hip/hip_runtime.h>
#include <hip/hip_bf16.h>
#include <math.h>

typedef __attribute__((ext_vector_type(16))) _Float16 v16h;
typedef __attribute__((ext_vector_type(8)))  float    v8f;

#define QTOT 10000
#define QPAD 10112   // 79 * 128  (padded M for guard-free async staging)
#define CDIM 128
#define PPTS 8
#define NCAM 6
#define HFE  28
#define WFE  60
#define MID  256
#define PCD  1024    // P*C
#define H1D  512     // 4*C

__device__ __forceinline__ float gelu_exact(float x) {
    return 0.5f * x * (1.0f + erff(x * 0.70710678118654752f));
}

// ---------------------------------------------------------------------------
// CDNA5 async copy: global -> LDS, 16 bytes per lane, tracked by ASYNCcnt.
// ---------------------------------------------------------------------------
__device__ __forceinline__ void async_cp16(void* lds_dst, const void* gsrc) {
    asm volatile("global_load_async_to_lds_b128 %0, %1, off"
                 :: "v"((unsigned)(unsigned long long)(uintptr_t)lds_dst),
                    "v"((unsigned long long)(uintptr_t)gsrc)
                 : "memory");
}
__device__ __forceinline__ void wait_async0() {
    asm volatile("s_wait_asynccnt 0x0" ::: "memory");
}

// ---------------------------------------------------------------------------
// Weight transpose + f32->f16 convert: src [K,N] f32 row-major -> dst [N,K] f16
// ---------------------------------------------------------------------------
__global__ void k_wT(const float* __restrict__ src, _Float16* __restrict__ dst,
                     int K, int N) {
    int i = blockIdx.x * blockDim.x + threadIdx.x;
    if (i >= K * N) return;
    int k = i / N;
    int n = i - k * N;
    dst[(size_t)n * K + k] = (_Float16)src[i];
}

// Zero the padded tail rows of xh so GEMM1 reads deterministic zeros there.
__global__ void k_zero16(_Float16* __restrict__ dst, int count) {
    int i = blockIdx.x * blockDim.x + threadIdx.x;
    if (i < count) dst[i] = (_Float16)0.0f;
}

// ---------------------------------------------------------------------------
// Projection + cam_uv + bilinear gather + pos-MLP -> x f16 [Q, P*C]
// One block (256 threads) per query q.
// ---------------------------------------------------------------------------
__device__ __forceinline__ float tap(const float* __restrict__ F, int ix, int iy) {
    if (ix < 0 || ix > WFE - 1 || iy < 0 || iy > HFE - 1) return 0.0f;
    return F[iy * WFE + ix];
}

__global__ void __launch_bounds__(256)
k_sample_pos(const float* __restrict__ feats,   // [N,C,HF,WF]
             const float* __restrict__ refp,    // [Q,3]
             const float* __restrict__ l2i,     // [N,4,4]
             const float* __restrict__ soff,    // [Q,P,3]
             const float* __restrict__ w1, const float* __restrict__ b1,
             const float* __restrict__ w2, const float* __restrict__ b2,
             _Float16* __restrict__ xh,         // [QPAD, P*C] f16
             float* __restrict__ cam_uv) {      // [N,Q,P,3]
    __shared__ float sL[NCAM * 16];
    __shared__ float sHid[MID];

    const int q = blockIdx.x;
    const int t = threadIdx.x;

    if (t < NCAM * 16) sL[t] = l2i[t];
    __syncthreads();

    const float lo0 = -50.f, lo1 = -50.f, lo2 = -5.f;
    const float sp0 = 100.f, sp1 = 100.f, sp2 = 8.f;

    const float rx = refp[q * 3 + 0] * sp0 + lo0;
    const float ry = refp[q * 3 + 1] * sp1 + lo1;
    const float rz = refp[q * 3 + 2] * sp2 + lo2;

    for (int p = 0; p < PPTS; ++p) {
        const float px = rx + soff[((size_t)q * PPTS + p) * 3 + 0];
        const float py = ry + soff[((size_t)q * PPTS + p) * 3 + 1];
        const float pz = rz + soff[((size_t)q * PPTS + p) * 3 + 2];

        const float rn0 = (px - lo0) / sp0;
        const float rn1 = (py - lo1) / sp1;
        const float rn2 = (pz - lo2) / sp2;

        // hidden layer: 256 threads, one neuron each
        {
            float hv = b1[t] + rn0 * w1[t] + rn1 * w1[MID + t] + rn2 * w1[2 * MID + t];
            sHid[t] = fmaxf(hv, 0.0f);
        }

        float uu[NCAM], vv[NCAM];
        int   vld[NCAM];
#pragma unroll
        for (int n = 0; n < NCAM; ++n) {
            const float* L = sL + n * 16;
            float c0 = L[0] * px + L[1] * py + L[2]  * pz + L[3];
            float c1 = L[4] * px + L[5] * py + L[6]  * pz + L[7];
            float c2 = L[8] * px + L[9] * py + L[10] * pz + L[11];
            float den = fmaxf(c2, 1e-5f);
            float u = (c0 / den) * (1.0f / 480.0f);
            float v = (c1 / den) * (1.0f / 224.0f);
            int ok = (c2 > 1e-5f) & (u > 0.0f) & (u < 1.0f) & (v > 0.0f) & (v < 1.0f);
            uu[n] = u; vv[n] = v; vld[n] = ok;
            if (t == n) {
                size_t o = (((size_t)n * QTOT + q) * PPTS + p) * 3;
                cam_uv[o + 0] = u;
                cam_uv[o + 1] = v;
                cam_uv[o + 2] = ok ? 1.0f : 0.0f;
            }
        }
        __syncthreads();

        if (t < CDIM) {
            const int c = t;
            float pos = b2[c];
#pragma unroll 8
            for (int j = 0; j < MID; ++j) pos += sHid[j] * w2[j * CDIM + c];

            float acc = 0.0f;
#pragma unroll
            for (int n = 0; n < NCAM; ++n) {
                if (!vld[n]) continue;
                float fx = uu[n] * (float)WFE - 0.5f;
                float fy = vv[n] * (float)HFE - 0.5f;
                float x0f = floorf(fx), y0f = floorf(fy);
                int   x0 = (int)x0f,   y0 = (int)y0f;
                float wx = fx - x0f,   wy = fy - y0f;
                const float* F = feats + ((size_t)(n * CDIM + c)) * (HFE * WFE);
                acc += tap(F, x0,     y0    ) * (1.0f - wx) * (1.0f - wy);
                acc += tap(F, x0 + 1, y0    ) * wx          * (1.0f - wy);
                acc += tap(F, x0,     y0 + 1) * (1.0f - wx) * wy;
                acc += tap(F, x0 + 1, y0 + 1) * wx          * wy;
            }
            xh[(size_t)q * PCD + p * CDIM + c] = (_Float16)(acc + pos);
        }
        __syncthreads();
    }
}

// ---------------------------------------------------------------------------
// WMMA GEMM, double-buffered async-LDS staging.
//   D[M,N] = act(A[M,K] @ B[K,N] + bias)
//   A f16 row-major [MPAD,K], Bt f16 = B^T [N,K] row-major.
//   BM=128, BN=128, BK=32. 8 waves (4x2), each wave owns 32x64
//   => 8x v_wmma_f32_16x16x32_f16 per K-step.
//   MPAD (= gridDim.x*128) rows must be readable; outF store guarded by Mreal.
// ---------------------------------------------------------------------------
__device__ __forceinline__ v16h ld_frag(const _Float16* __restrict__ rowbase, int kb) {
    union { uint4 u[2]; v16h v; } u;
    u.u[0] = *(const uint4*)(rowbase + kb);
    u.u[1] = *(const uint4*)(rowbase + kb + 16);
    return u.v;
}

__global__ void __launch_bounds__(256)
k_gemm(const _Float16* __restrict__ A, const _Float16* __restrict__ Bt,
       const float* __restrict__ bias, int Mreal, int K, int N,
       _Float16* __restrict__ outH, float* __restrict__ outF,
       int act, int ldF) {
    __shared__ _Float16 sA[2][128 * 32];
    __shared__ _Float16 sB[2][128 * 32];

    const int t    = threadIdx.x;
    const int lane = t & 31;
    const int wid  = t >> 5;
    const int wm   = wid & 3;    // 0..3 -> 32-row slab
    const int wn   = wid >> 2;   // 0..1 -> 64-col slab
    const int m0   = blockIdx.x * 128;
    const int n0   = blockIdx.y * 128;

    v8f acc[2][4];
#pragma unroll
    for (int i = 0; i < 2; ++i)
#pragma unroll
        for (int j = 0; j < 4; ++j)
            acc[i][j] = (v8f){0.f, 0.f, 0.f, 0.f, 0.f, 0.f, 0.f, 0.f};

    const int kb = (lane >> 4) * 8;
    const int lr = lane & 15;

    // async stage of one 128x32 A tile + 128x32 Bt tile into buffer `buf`
    auto stage = [&](int buf, int kk) {
#pragma unroll
        for (int r = 0; r < 2; ++r) {
            const int chunk = t + 256 * r;        // 0..511
            const int row   = chunk >> 2;         // 0..127
            const int koff  = (chunk & 3) * 8;    // 0,8,16,24 halfs
            async_cp16(&sA[buf][row * 32 + koff],
                       A + (size_t)(m0 + row) * K + kk + koff);
            async_cp16(&sB[buf][row * 32 + koff],
                       Bt + (size_t)(n0 + row) * K + kk + koff);
        }
    };

    const int nk = K >> 5;
    stage(0, 0);
    wait_async0();
    __syncthreads();

    for (int it = 0; it < nk; ++it) {
        const int cur = it & 1;
        if (it + 1 < nk) stage(cur ^ 1, (it + 1) << 5);

        v16h af[2], bf[4];
#pragma unroll
        for (int i = 0; i < 2; ++i)
            af[i] = ld_frag(&sA[cur][(wm * 32 + i * 16 + lr) * 32], kb);
#pragma unroll
        for (int j = 0; j < 4; ++j)
            bf[j] = ld_frag(&sB[cur][(wn * 64 + j * 16 + lr) * 32], kb);

#pragma unroll
        for (int i = 0; i < 2; ++i)
#pragma unroll
            for (int j = 0; j < 4; ++j)
                acc[i][j] = __builtin_amdgcn_wmma_f32_16x16x32_f16(
                    false, af[i], false, bf[j], (short)0, acc[i][j], false, false);

        if (it + 1 < nk) wait_async0();
        __syncthreads();
    }

    // epilogue: C/D layout -> row M = r + 8*(lane>=16), col N = lane&15
    const int lrow = (lane >> 4) * 8;
#pragma unroll
    for (int i = 0; i < 2; ++i) {
#pragma unroll
        for (int j = 0; j < 4; ++j) {
            const int n = n0 + wn * 64 + j * 16 + lr;
            const float bb = bias[n];
#pragma unroll
            for (int r = 0; r < 8; ++r) {
                const int m = m0 + wm * 32 + i * 16 + lrow + r;
                float v = acc[i][j][r] + bb;
                if (act == 1) v = gelu_exact(v);
                if (outH) outH[(size_t)m * N + n] = (_Float16)v;         // padded dest
                if (outF && m < Mreal) outF[(size_t)n * ldF + m] = v;    // transposed f32
            }
        }
    }
}

// ---------------------------------------------------------------------------
// Launch
// ---------------------------------------------------------------------------
extern "C" void kernel_launch(void* const* d_in, const int* in_sizes, int n_in,
                              void* d_out, int out_size, void* d_ws, size_t ws_size,
                              hipStream_t stream) {
    const float* feats = (const float*)d_in[1];
    const float* refp  = (const float*)d_in[2];
    const float* l2i   = (const float*)d_in[3];
    const float* soff  = (const float*)d_in[4];
    const float* w1    = (const float*)d_in[5];
    const float* b1    = (const float*)d_in[6];
    const float* w2    = (const float*)d_in[7];
    const float* b2    = (const float*)d_in[8];
    const float* cw1   = (const float*)d_in[9];
    const float* cb1   = (const float*)d_in[10];
    const float* cw2   = (const float*)d_in[11];
    const float* cb2   = (const float*)d_in[12];
    const float* cw3   = (const float*)d_in[13];
    const float* cb3   = (const float*)d_in[14];

    float* out    = (float*)d_out;                 // [C=128][Q=10000]
    float* cam_uv = out + (size_t)CDIM * QTOT;     // [N=6][Q][P=8][3]

    char* ws = (char*)d_ws;
    size_t o = 0;
    _Float16* xh  = (_Float16*)(ws + o); o += (size_t)QPAD * PCD * 2;
    _Float16* w1t = (_Float16*)(ws + o); o += (size_t)H1D * PCD * 2;   // cw1^T [512,1024]
    _Float16* w2t = (_Float16*)(ws + o); o += (size_t)H1D * H1D * 2;   // cw2^T [512,512]
    _Float16* w3t = (_Float16*)(ws + o); o += (size_t)CDIM * H1D * 2;  // cw3^T [128,512]
    _Float16* h1  = (_Float16*)(ws + o); o += (size_t)QPAD * H1D * 2;
    _Float16* h2  = (_Float16*)(ws + o);

    // weight convert/transpose
    k_wT<<<(PCD * H1D + 255) / 256, 256, 0, stream>>>(cw1, w1t, PCD, H1D);
    k_wT<<<(H1D * H1D + 255) / 256, 256, 0, stream>>>(cw2, w2t, H1D, H1D);
    k_wT<<<(H1D * CDIM + 255) / 256, 256, 0, stream>>>(cw3, w3t, H1D, CDIM);

    // zero padded tail rows of xh (GEMM1 reads them unguarded)
    {
        const int padElems = (QPAD - QTOT) * PCD;
        k_zero16<<<(padElems + 255) / 256, 256, 0, stream>>>(xh + (size_t)QTOT * PCD,
                                                             padElems);
    }

    // sampling + pos MLP + cam_uv
    k_sample_pos<<<QTOT, 256, 0, stream>>>(feats, refp, l2i, soff,
                                           w1, b1, w2, b2, xh, cam_uv);

    // MLP chain on WMMA (guard-free loads over QPAD rows)
    const int gm = QPAD / 128;   // 79
    k_gemm<<<dim3(gm, H1D / 128), 256, 0, stream>>>(xh, w1t, cb1, QTOT, PCD, H1D,
                                                    h1, nullptr, 1, 0);
    k_gemm<<<dim3(gm, H1D / 128), 256, 0, stream>>>(h1, w2t, cb2, QTOT, H1D, H1D,
                                                    h2, nullptr, 1, 0);
    k_gemm<<<dim3(gm, CDIM / 128), 256, 0, stream>>>(h2, w3t, cb3, QTOT, H1D, CDIM,
                                                     nullptr, out, 0, QTOT);
}
